// MultiHeadSelfAttention_21964462752409
// MI455X (gfx1250) — matrix operand order repaired and checked by reading, not compiled, not run on hardware
//
#include <hip/hip_runtime.h>
#include <hip/hip_bf16.h>

// ---------------------------------------------------------------------------
// MHSA for MI455X (gfx1250, wave32, WMMA). f16 multiplicands, f32 accumulate.
// Register-blocked WMMA GEMMs: 2x4 tiles/wave for projections, 1x4 for
// scores and ctx (ctx reads the 537MB attn matrix exactly once).
// ---------------------------------------------------------------------------

typedef __attribute__((ext_vector_type(16))) _Float16 v16h;
typedef __attribute__((ext_vector_type(8)))  _Float16 v8h;
typedef __attribute__((ext_vector_type(8)))  float    v8f;
typedef __attribute__((ext_vector_type(4)))  float    v4f;

#define Bsz 8
#define Nseq 1024
#define Hdim 1024
#define NH 16
#define HD 64

// ---------------- fragment loaders (ISA 7.12.2 layouts, wave32) -------------

// A matrix 16x32 f16, row-major source with leading dim ld.
// lane: m = lane&15 (row), h = lane>>4. elems 0..7 -> K = h*8+e ; 8..15 -> K = 16+h*8+e
__device__ __forceinline__ v16h load_a_f16(const _Float16* __restrict__ rowbase,
                                           int ld, int kb, int lane) {
    int m = lane & 15, h = lane >> 4;
    const _Float16* p = rowbase + (size_t)m * ld + kb + h * 8;
    v8h lo = *(const v8h*)p;
    v8h hi = *(const v8h*)(p + 16);
    v16h r;
#pragma unroll
    for (int e = 0; e < 8; ++e) { r[e] = lo[e]; r[e + 8] = hi[e]; }
    return r;
}

// B matrix 32x16, but source is B^T row-major: TB[n][k], leading dim ld.
// lane: n = lane&15 (output col), h = lane>>4. elems 0..7 -> K = h*8+e ; 8..15 -> K = 16+h*8+e
__device__ __forceinline__ v16h load_bT_f16(const _Float16* __restrict__ colbase,
                                            int ld, int kb, int lane) {
    int n = lane & 15, h = lane >> 4;
    const _Float16* p = colbase + (size_t)n * ld + kb + h * 8;
    v8h lo = *(const v8h*)p;
    v8h hi = *(const v8h*)(p + 16);
    v16h r;
#pragma unroll
    for (int e = 0; e < 8; ++e) { r[e] = lo[e]; r[e + 8] = hi[e]; }
    return r;
}

// A matrix 16x32 from f32 source (attention probabilities), convert to f16.
__device__ __forceinline__ v16h load_a_f32(const float* __restrict__ rowbase,
                                           int ld, int kb, int lane) {
    int m = lane & 15, h = lane >> 4;
    const float* p = rowbase + (size_t)m * ld + kb + h * 8;
    v4f a0 = *(const v4f*)p;
    v4f a1 = *(const v4f*)(p + 4);
    v4f a2 = *(const v4f*)(p + 16);
    v4f a3 = *(const v4f*)(p + 20);
    v16h r;
#pragma unroll
    for (int e = 0; e < 4; ++e) {
        r[e]      = (_Float16)a0[e];
        r[4 + e]  = (_Float16)a1[e];
        r[8 + e]  = (_Float16)a2[e];
        r[12 + e] = (_Float16)a3[e];
    }
    return r;
}

__device__ __forceinline__ v8f wmma_f16(v16h a, v16h b, v8f c) {
    return __builtin_amdgcn_wmma_f32_16x16x32_f16(false, a, false, b,
                                                  (short)0, c, false, false);
}

// ---------------- conversion kernels ---------------------------------------

__global__ __launch_bounds__(256) void cvt_f32_to_f16_kernel(
    const float* __restrict__ src, _Float16* __restrict__ dst, int n) {
    int i = blockIdx.x * 256 + threadIdx.x;
    if (i < n) dst[i] = (_Float16)src[i];
}

// wT[n][k] = (f16) w[k][n], w is Hdim x Hdim
__global__ __launch_bounds__(256) void transpose_cvt_kernel(
    const float* __restrict__ w, _Float16* __restrict__ wT) {
    int i = blockIdx.x * 256 + threadIdx.x;   // over Hdim*Hdim
    int n = i & (Hdim - 1);
    int k = i >> 10;
    wT[(size_t)n * Hdim + k] = (_Float16)w[(size_t)k * Hdim + n];
}

// ---------------- generic WMMA GEMM: C = A(f16) x BT(f16)^T + bias ---------
// A: [M x K] row-major.  BT: [Ncols x K] row-major (i.e. B transposed).
// Each wave computes a 32(M) x 64(N) block = 2x4 WMMA tiles.
// Block = 4 waves covering 32 x 256.  Grid: (Ncols/256, M/32).
// mode 0: f16 dest, head layout [B,NH,N,HD]
// mode 1: f16 dest, head-transposed layout [B,NH,HD,N]
// mode 2: f32 dest, row-major [M x Ncols]
__global__ __launch_bounds__(128) void gemm_f16_kernel(
    const _Float16* __restrict__ A, const _Float16* __restrict__ BT,
    const float* __restrict__ bias, void* __restrict__ dst,
    int mode, int K, int Ncols) {
    int lane = threadIdx.x & 31;
    int wave = threadIdx.x >> 5;
    int i0 = blockIdx.y * 32;
    int j0 = (blockIdx.x * 4 + wave) * 64;

    const _Float16* arow0 = A + (size_t)i0 * K;
    const _Float16* arow1 = A + (size_t)(i0 + 16) * K;

    v8f c[2][4];
    v8f z = {};
#pragma unroll
    for (int ti = 0; ti < 2; ++ti)
#pragma unroll
        for (int tj = 0; tj < 4; ++tj) c[ti][tj] = z;

    for (int kb = 0; kb < K; kb += 32) {
        __builtin_prefetch((const void*)(arow0 + (size_t)(lane & 15) * K + kb + 64), 0, 0);
        v16h a0 = load_a_f16(arow0, K, kb, lane);
        v16h a1 = load_a_f16(arow1, K, kb, lane);
#pragma unroll
        for (int tj = 0; tj < 4; ++tj) {
            v16h b = load_bT_f16(BT + (size_t)(j0 + tj * 16) * K, K, kb, lane);
            c[0][tj] = wmma_f16(a0, b, c[0][tj]);
            c[1][tj] = wmma_f16(a1, b, c[1][tj]);
        }
    }

    int n = lane & 15, h = lane >> 4;
#pragma unroll
    for (int ti = 0; ti < 2; ++ti) {
#pragma unroll
        for (int tj = 0; tj < 4; ++tj) {
            int gcb = j0 + tj * 16 + n;
            float bv = bias ? bias[gcb] : 0.0f;
#pragma unroll
            for (int r = 0; r < 8; ++r) {
                int gm = i0 + ti * 16 + r + 8 * h;   // global row in [0, B*N)
                float val = c[ti][tj][r] + bv;
                if (mode == 0) {
                    int b_ = gm >> 10, nn = gm & 1023, hh = gcb >> 6, d = gcb & 63;
                    ((_Float16*)dst)[((size_t)(b_ * NH + hh) * Nseq + nn) * HD + d] =
                        (_Float16)val;
                } else if (mode == 1) {
                    int b_ = gm >> 10, nn = gm & 1023, hh = gcb >> 6, d = gcb & 63;
                    ((_Float16*)dst)[((size_t)(b_ * NH + hh) * HD + d) * Nseq + nn] =
                        (_Float16)val;
                } else {
                    ((float*)dst)[(size_t)gm * Ncols + gcb] = val;
                }
            }
        }
    }
}

// ---------------- scores: S = Q Kh^T / sqrt(HD) -----------------------------
// qh, kh: [B,NH,N,HD] f16. attn: [B,NH,N,N] f32 (raw scores written here).
// Each wave: 16(q) x 64(k) block = 1x4 tiles.  Grid: (N/256, N/16, B*NH).
__global__ __launch_bounds__(128) void scores_kernel(
    const _Float16* __restrict__ qh, const _Float16* __restrict__ kh,
    float* __restrict__ attn) {
    int lane = threadIdx.x & 31;
    int wave = threadIdx.x >> 5;
    int bh = blockIdx.z;
    int i0 = blockIdx.y * 16;
    int j0 = (blockIdx.x * 4 + wave) * 64;

    const _Float16* q = qh + (size_t)bh * Nseq * HD;
    const _Float16* k = kh + (size_t)bh * Nseq * HD;

    v8f c[4];
    v8f z = {};
#pragma unroll
    for (int tj = 0; tj < 4; ++tj) c[tj] = z;

#pragma unroll
    for (int kb = 0; kb < HD; kb += 32) {
        v16h a = load_a_f16(q + (size_t)i0 * HD, HD, kb, lane);
#pragma unroll
        for (int tj = 0; tj < 4; ++tj) {
            v16h b = load_bT_f16(k + (size_t)(j0 + tj * 16) * HD, HD, kb, lane);
            c[tj] = wmma_f16(a, b, c[tj]);
        }
    }

    float* outp = attn + (size_t)bh * Nseq * Nseq;
    int n = lane & 15, h = lane >> 4;
#pragma unroll
    for (int tj = 0; tj < 4; ++tj) {
#pragma unroll
        for (int r = 0; r < 8; ++r) {
            outp[(size_t)(i0 + r + 8 * h) * Nseq + j0 + tj * 16 + n] =
                c[tj][r] * 0.125f;  // 1/sqrt(64)
        }
    }
}

// ---------------- row softmax, in place on attn -----------------------------
__global__ __launch_bounds__(256) void softmax_kernel(float* __restrict__ attn) {
    __shared__ float red[256];
    int t = threadIdx.x;
    float* row = attn + (size_t)blockIdx.x * Nseq;

    float v0 = row[t], v1 = row[t + 256], v2 = row[t + 512], v3 = row[t + 768];
    float m = fmaxf(fmaxf(v0, v1), fmaxf(v2, v3));
    red[t] = m;
    __syncthreads();
    for (int s = 128; s > 0; s >>= 1) {
        if (t < s) red[t] = fmaxf(red[t], red[t + s]);
        __syncthreads();
    }
    float mx = red[0];
    __syncthreads();

    float e0 = __expf(v0 - mx), e1 = __expf(v1 - mx);
    float e2 = __expf(v2 - mx), e3 = __expf(v3 - mx);
    red[t] = e0 + e1 + e2 + e3;
    __syncthreads();
    for (int s = 128; s > 0; s >>= 1) {
        if (t < s) red[t] += red[t + s];
        __syncthreads();
    }
    float inv = 1.0f / red[0];
    row[t] = e0 * inv; row[t + 256] = e1 * inv;
    row[t + 512] = e2 * inv; row[t + 768] = e3 * inv;
}

// ---------------- ctx = P x V  (P f32 -> f16 on load; V pre-transposed) ----
// attn: [B,NH,N,N] f32.  vT: [B,NH,HD,N] f16.  ctx16: [B*N, H] f16 row-major.
// Each wave: 16 query rows x all 64 hd cols = 1x4 tiles -> attn read ONCE.
// Grid: (N/64, B*NH), block = 4 waves.
__global__ __launch_bounds__(128) void ctx_kernel(
    const float* __restrict__ attn, const _Float16* __restrict__ vT,
    _Float16* __restrict__ ctx16) {
    int lane = threadIdx.x & 31;
    int wave = threadIdx.x >> 5;
    int bh = blockIdx.y;
    int b_ = bh >> 4, hh = bh & 15;
    int i0 = (blockIdx.x * 4 + wave) * 16;

    const float* P = attn + (size_t)bh * Nseq * Nseq;
    const _Float16* V = vT + (size_t)bh * HD * Nseq;

    v8f c[4];
    v8f z = {};
#pragma unroll
    for (int tj = 0; tj < 4; ++tj) c[tj] = z;

    for (int kb = 0; kb < Nseq; kb += 32) {
        v16h a = load_a_f32(P + (size_t)i0 * Nseq, Nseq, kb, lane);
#pragma unroll
        for (int tj = 0; tj < 4; ++tj) {
            v16h b = load_bT_f16(V + (size_t)(tj * 16) * Nseq, Nseq, kb, lane);
            c[tj] = wmma_f16(a, b, c[tj]);
        }
    }

    int n = lane & 15, h = lane >> 4;
#pragma unroll
    for (int tj = 0; tj < 4; ++tj) {
#pragma unroll
        for (int r = 0; r < 8; ++r) {
            int nq = i0 + r + 8 * h;
            ctx16[((size_t)(b_ * Nseq + nq)) * Hdim + hh * HD + tj * 16 + n] =
                (_Float16)c[tj][r];
        }
    }
}

// ---------------------------------------------------------------------------

extern "C" void kernel_launch(void* const* d_in, const int* in_sizes, int n_in,
                              void* d_out, int out_size, void* d_ws, size_t ws_size,
                              hipStream_t stream) {
    const float* x  = (const float*)d_in[0];
    const float* wq = (const float*)d_in[1];
    const float* bq = (const float*)d_in[2];
    const float* wk = (const float*)d_in[3];
    const float* bk = (const float*)d_in[4];
    const float* wv = (const float*)d_in[5];
    const float* bv = (const float*)d_in[6];
    const float* wo = (const float*)d_in[7];
    const float* bo = (const float*)d_in[8];

    float* out  = (float*)d_out;                                  // [B,N,H]
    float* attn = out + (size_t)Bsz * Nseq * Hdim;                // [B,NH,N,N]

    const size_t NE_X = (size_t)Bsz * Nseq * Hdim;   // 8,388,608
    const size_t NE_W = (size_t)Hdim * Hdim;         // 1,048,576

    _Float16* ws   = (_Float16*)d_ws;
    _Float16* x16  = ws;
    _Float16* wqT  = x16 + NE_X;
    _Float16* wkT  = wqT + NE_W;
    _Float16* wvT  = wkT + NE_W;
    _Float16* woT  = wvT + NE_W;
    _Float16* qh   = woT + NE_W;
    _Float16* kh   = qh + NE_X;
    _Float16* vT   = kh + NE_X;
    _Float16* c16  = vT + NE_X;

    const int M = Bsz * Nseq;   // 8192

    // 1) precision staging
    cvt_f32_to_f16_kernel<<<(int)(NE_X / 256), 256, 0, stream>>>(x, x16, (int)NE_X);
    transpose_cvt_kernel<<<(int)(NE_W / 256), 256, 0, stream>>>(wq, wqT);
    transpose_cvt_kernel<<<(int)(NE_W / 256), 256, 0, stream>>>(wk, wkT);
    transpose_cvt_kernel<<<(int)(NE_W / 256), 256, 0, stream>>>(wv, wvT);
    transpose_cvt_kernel<<<(int)(NE_W / 256), 256, 0, stream>>>(wo, woT);

    // 2) QKV projections (WMMA): 2x4 tiles/wave, grid (Ncols/256, M/32)
    dim3 gemm_grid(Hdim / 256, M / 32, 1);
    gemm_f16_kernel<<<gemm_grid, 128, 0, stream>>>(x16, wqT, bq, qh, 0, Hdim, Hdim);
    gemm_f16_kernel<<<gemm_grid, 128, 0, stream>>>(x16, wkT, bk, kh, 0, Hdim, Hdim);
    gemm_f16_kernel<<<gemm_grid, 128, 0, stream>>>(x16, wvT, bv, vT, 1, Hdim, Hdim);

    // 3) scores -> attn (raw, scaled): 1x4 tiles/wave
    dim3 sc_grid(Nseq / 256, Nseq / 16, Bsz * NH);
    scores_kernel<<<sc_grid, 128, 0, stream>>>(qh, kh, attn);

    // 4) softmax in place (one block per row)
    softmax_kernel<<<Bsz * NH * Nseq, 256, 0, stream>>>(attn);

    // 5) ctx = P @ V : 1x4 tiles/wave, attn read exactly once
    dim3 ctx_grid(Nseq / 64, Bsz * NH, 1);
    ctx_kernel<<<ctx_grid, 128, 0, stream>>>(attn, vT, c16);

    // 6) output projection -> f32 out
    gemm_f16_kernel<<<gemm_grid, 128, 0, stream>>>(c16, woT, bo, out, 2, Hdim, Hdim);
}
